// DotproductAttention_71760313582141
// MI455X (gfx1250) — compile-verified, matrix operand order checked by reading, not run on hardware
//
#include <hip/hip_runtime.h>
#include <hip/hip_bf16.h>

typedef __attribute__((ext_vector_type(16))) _Float16 v16h;
typedef __attribute__((ext_vector_type(8)))  _Float16 v8h;
typedef __attribute__((ext_vector_type(4)))  _Float16 v4h;
typedef __attribute__((ext_vector_type(8)))  float    v8f;
typedef __attribute__((ext_vector_type(4)))  float    v4f;

#define NEGVAL (-1000000.0f)

constexpr int Bn      = 16;
constexpr int Sq      = 2048;
constexpr int Sk      = 2048;
constexpr int Dh      = 128;
constexpr int QTILE   = 128;   // queries per block: 8 waves * 16 rows
constexpr int KTILE   = 32;    // keys per mainloop iteration
constexpr int VSTRIDE = 40;    // padded key-stride (halfs) of transposed V tile; 80B, 16B-aligned

// Flash-attention, wave32 WMMA formulation:
//   GEMM1: S^T = K_tile x Q^T ; GEMM2: O^T += V^T x P^T
// S^T accumulator layout feeds the P^T B-fragment with zero cross-lane moves.
// Double-buffered LDS; global loads for tile t+1 issue before tile t's WMMAs.
// Fragment LDS reads are software-pipelined one step ahead of the consuming
// WMMA so no wmma waits on the loads issued directly before it.
__global__ __launch_bounds__(256)
void fa_fwd_kernel(const float* __restrict__ qg,
                   const float* __restrict__ kg,
                   const float* __restrict__ vg,
                   const int*   __restrict__ valid_len,
                   float*       __restrict__ outg)
{
    __shared__ _Float16 Ksh [2][KTILE * Dh];      // [buf][key][d]   2 x 8 KB
    __shared__ _Float16 VshT[2][Dh * VSTRIDE];    // [buf][d][key]   2 x 10 KB

    const int tid  = threadIdx.x;
    const int lane = tid & 31;
    const int wave = tid >> 5;
    const int ln   = lane & 15;    // query column / matrix row-in-lane
    const int hi   = lane >> 4;    // half-wave select

    const int b      = blockIdx.y;
    const int qBlock = blockIdx.x * QTILE;
    const int qRow   = qBlock + wave * 16 + ln;
    const int vlen   = valid_len[b];

    const float* kb = kg + (size_t)b * Sk * Dh;
    const float* vb = vg + (size_t)b * Sk * Dh;

    // ---- vlen==0: softmax over a constant row -> uniform weights -> mean(V) ----
    if (vlen == 0) {
        float* red = reinterpret_cast<float*>(&Ksh[0][0]);   // 256 floats scratch
        const int d  = tid & 127;
        const int h2 = tid >> 7;
        float acc = 0.0f;
        for (int s = h2; s < Sk; s += 2) acc += vb[(size_t)s * Dh + d];
        red[tid] = acc;
        __syncthreads();
        const float mean = (red[d] + red[d + 128]) * (1.0f / (float)Sk);
        for (int r = h2; r < QTILE; r += 2)
            outg[((size_t)b * Sq + qBlock + r) * Dh + d] = mean;
        return;
    }

    const float sscale = 0.08838834764831845f;  // 1/sqrt(128), folded into Q

    // ---- Q^T B-fragments (pre-scaled), resident for the whole kernel ----
    v16h bq[4];
    {
        const float* qp = qg + ((size_t)b * Sq + qRow) * Dh;
#pragma unroll
        for (int c = 0; c < 4; ++c) {
            const float* p0 = qp + 32 * c + hi * 8;        // halves 0..7
            const float* p1 = qp + 32 * c + 16 + hi * 8;   // halves 8..15
            v16h t;
#pragma unroll
            for (int j = 0; j < 8; ++j) {
                t[j]     = (_Float16)(p0[j] * sscale);
                t[j + 8] = (_Float16)(p1[j] * sscale);
            }
            bq[c] = t;
        }
    }

    // O^T accumulators: 8 fragments cover d = 0..127 (16 per fragment)
    v8f o[8] = {};
    float m_i = -INFINITY;
    float l_i = 0.0f;

    const int nTiles = ((vlen + KTILE - 1) & ~(KTILE - 1)) / KTILE;   // >= 1

    // register staging for the double-buffer pipeline
    v4f kreg[4], vreg[4];

    auto loadTile = [&](int kt) {
#pragma unroll
        for (int p = 0; p < 4; ++p) {
            const int i = tid + p * 256;
            kreg[p] = *(const v4f*)(kb + (size_t)kt * Dh + i * 4);
            vreg[p] = *(const v4f*)(vb + (size_t)kt * Dh + i * 4);
        }
    };
    auto storeTile = [&](int buf) {
#pragma unroll
        for (int p = 0; p < 4; ++p) {
            const int i   = tid + p * 256;
            const int key = (i * 4) >> 7;     // 0..31
            const int col = (i * 4) & 127;    // d, multiple of 4
            v4h kh;
#pragma unroll
            for (int j = 0; j < 4; ++j) kh[j] = (_Float16)kreg[p][j];
            *(v4h*)(&Ksh[buf][i * 4]) = kh;
#pragma unroll
            for (int j = 0; j < 4; ++j)
                VshT[buf][(col + j) * VSTRIDE + key] = (_Float16)vreg[p][j];
        }
    };

    loadTile(0);
    storeTile(0);

    for (int t = 0; t < nTiles; ++t) {
        const int kt  = t * KTILE;
        const int cur = t & 1;
        __syncthreads();                       // buffer `cur` ready, other buffer free

        if (t + 1 < nTiles) loadTile(kt + KTILE);   // HBM latency hidden by WMMAs below

        const _Float16* kbase = &Ksh[cur][0];
        const _Float16* vbase = &VshT[cur][0];

        auto ldA = [&](int row16, int c, v16h& a) {   // K-tile A fragment
            const _Float16* r = kbase + (row16 + ln) * Dh + 32 * c + hi * 8;
            const v8h x = *(const v8h*)r;
            const v8h y = *(const v8h*)(r + 16);
#pragma unroll
            for (int j = 0; j < 8; ++j) { a[j] = x[j]; a[j + 8] = y[j]; }
        };
        auto ldV = [&](int n, v16h& a) {              // V^T A fragment (d-chunk n)
            const _Float16* vc = vbase + (16 * n + ln) * VSTRIDE;
            const v8h x = *(const v8h*)(vc + hi * 8);
            const v8h y = *(const v8h*)(vc + 16 + hi * 8);
#pragma unroll
            for (int j = 0; j < 8; ++j) { a[j] = x[j]; a[j + 8] = y[j]; }
        };

        // ---- GEMM1: S^T = K_tile . (Q*scale)^T, fragment loads pipelined ----
        v8f s0 = {}, s1 = {};
        {
            v16h a0, a1;
            ldA(0, 0, a0);
            ldA(16, 0, a1);
#pragma unroll
            for (int c = 0; c < 4; ++c) {
                v16h n0, n1;
                if (c < 3) { ldA(0, c + 1, n0); ldA(16, c + 1, n1); }
                s0 = __builtin_amdgcn_wmma_f32_16x16x32_f16(false, a0, false, bq[c],
                                                            (short)0, s0, false, false);
                s1 = __builtin_amdgcn_wmma_f32_16x16x32_f16(false, a1, false, bq[c],
                                                            (short)0, s1, false, false);
                if (c < 3) { a0 = n0; a1 = n1; }
            }
        }

        // ---- mask only the (single) partial tail tile ----
        if (kt + KTILE > vlen) {
            const int rem = vlen - kt - 8 * hi;    // per-lane; key0 = kt+8hi+j
#pragma unroll
            for (int j = 0; j < 8; ++j) {
                if (j >= rem)      s0[j] = NEGVAL;  // kt+8hi+j    >= vlen
                if (j >= rem - 16) s1[j] = NEGVAL;  // kt+16+8hi+j >= vlen
            }
        }

        // prefetch first V^T fragment; its LDS latency hides under the softmax
        v16h av;
        ldV(0, av);

        // ---- online softmax (per-query state per-lane; halves hold same query) ----
        float tmax = s0[0];
#pragma unroll
        for (int j = 1; j < 8; ++j) tmax = fmaxf(tmax, s0[j]);
#pragma unroll
        for (int j = 0; j < 8; ++j) tmax = fmaxf(tmax, s1[j]);
        tmax = fmaxf(tmax, __shfl_xor(tmax, 16, 32));

        float corr = 1.0f;
        if (__ballot(tmax > m_i)) {            // wave-uniform: skip exact-identity rescale
            const float mNew = fmaxf(m_i, tmax);
            corr = __expf(m_i - mNew);
            m_i  = mNew;
#pragma unroll
            for (int n = 0; n < 8; ++n)
#pragma unroll
                for (int j = 0; j < 8; ++j) o[n][j] *= corr;
        }

        v16h pT;
        float lsum = 0.0f;
#pragma unroll
        for (int j = 0; j < 8; ++j) {
            const float p0 = __expf(s0[j] - m_i);
            const float p1 = __expf(s1[j] - m_i);
            pT[j]     = (_Float16)p0;
            pT[j + 8] = (_Float16)p1;
            lsum += p0 + p1;
        }
        lsum += __shfl_xor(lsum, 16, 32);
        l_i = l_i * corr + lsum;

        // ---- GEMM2: O^T += V^T . P^T, fragment loads pipelined ----
#pragma unroll
        for (int n = 0; n < 8; ++n) {
            v16h avn;
            if (n < 7) ldV(n + 1, avn);
            o[n] = __builtin_amdgcn_wmma_f32_16x16x32_f16(false, av, false, pT,
                                                          (short)0, o[n], false, false);
            if (n < 7) av = avn;
        }

        if (t + 1 < nTiles) storeTile(1 - cur);     // convert + stage next buffer
    }

    // ---- normalize + write out (vectorized 16B stores) ----
    const float invl = 1.0f / l_i;
    float* op = outg + ((size_t)b * Sq + qRow) * Dh + 8 * hi;
#pragma unroll
    for (int n = 0; n < 8; ++n) {
        v4f lo4, hi4;
#pragma unroll
        for (int j = 0; j < 4; ++j) {
            lo4[j] = o[n][j]     * invl;
            hi4[j] = o[n][4 + j] * invl;
        }
        *(v4f*)(op + 16 * n)     = lo4;
        *(v4f*)(op + 16 * n + 4) = hi4;
    }
}

extern "C" void kernel_launch(void* const* d_in, const int* in_sizes, int n_in,
                              void* d_out, int out_size, void* d_ws, size_t ws_size,
                              hipStream_t stream) {
    const float* q  = (const float*)d_in[0];
    const float* k  = (const float*)d_in[1];
    const float* v  = (const float*)d_in[2];
    const int*   vl = (const int*)d_in[3];
    float* out = (float*)d_out;

    dim3 grid(Sq / QTILE, Bn, 1);   // 16 x 16 blocks
    fa_fwd_kernel<<<grid, 256, 0, stream>>>(q, k, v, vl, out);
}
